// FusedAttentionMoELayer_65369402245587
// MI455X (gfx1250) — matrix operand order repaired and checked by reading, not compiled, not run on hardware
//
#include <hip/hip_runtime.h>

// ---------------- problem constants ----------------
static constexpr int   kB     = 4;
static constexpr int   kS     = 1024;
static constexpr int   kH     = 1024;
static constexpr int   kNH    = 16;
static constexpr int   kHD    = 64;
static constexpr int   kINTER = 2048;
static constexpr int   kE     = 8;
static constexpr int   kT     = kB * kS;                 // 4096 tokens
static constexpr int   kQKVN  = (kNH + 2 * kNH) * kHD;   // 3072
static constexpr int   kNA    = 2 * kT + kE * 64;        // 8704 padded assignment rows
static constexpr int   kMAXT  = 160;                     // grouped-gemm row-tile capacity
static constexpr float kEPS   = 1e-6f;

// ---------------- types ----------------
typedef __bf16 bf16;
typedef __attribute__((ext_vector_type(16))) __bf16 v16bf;
typedef __attribute__((ext_vector_type(8)))  float  v8f;
typedef __attribute__((ext_vector_type(4)))  unsigned int v4u;
typedef __attribute__((ext_vector_type(8)))  int  v8i;
typedef __attribute__((ext_vector_type(4)))  int  v4i;

union V16 { uint4 q[2]; v16bf v; };

__device__ inline bf16 f2bf(float f) {
  unsigned u = __builtin_bit_cast(unsigned, f);
  unsigned r = u + 0x7FFFu + ((u >> 16) & 1u);
  unsigned short h = (unsigned short)(r >> 16);
  return __builtin_bit_cast(bf16, h);
}
__device__ inline float bf2f(bf16 b) {
  unsigned short h = __builtin_bit_cast(unsigned short, b);
  return __builtin_bit_cast(float, (unsigned)h << 16);
}
__device__ inline v16bf ld_frag(const bf16* p0, const bf16* p1) {
  V16 t;
  t.q[0] = *(const uint4*)p0;
  t.q[1] = *(const uint4*)p1;
  return t.v;
}
__device__ inline v8f wmma_bf16(v16bf a, v16bf b, v8f c) {
  return __builtin_amdgcn_wmma_f32_16x16x32_bf16(false, a, false, b, (short)0, c,
                                                 false, false);
}
__device__ inline void wavebar() {
#if __has_builtin(__builtin_amdgcn_wave_barrier)
  __builtin_amdgcn_wave_barrier();
#endif
}
__device__ inline float redmax16(float v) {
  for (int off = 1; off < 16; off <<= 1) v = fmaxf(v, __shfl_xor(v, off));
  return v;
}
__device__ inline float redsum16(float v) {
  for (int off = 1; off < 16; off <<= 1) v += __shfl_xor(v, off);
  return v;
}

// ---------------- Tensor Data Mover (gfx1250 TDM) ----------------
#if __has_builtin(__builtin_amdgcn_tensor_load_to_lds) && \
    __has_builtin(__builtin_amdgcn_s_wait_tensorcnt)
#define USE_TDM 1
#endif

#ifdef USE_TDM
// 2D bf16 tile DMA: `rows` rows of `cols` elements, global row stride
// `stride_elems`. LDS destination rows padded by the TDM itself:
// padi: pad-interval code (2^(padi+1) dwords), pada: pad-amount code (pada+1 dwords).
__device__ inline void tdm_load_2d(const bf16* gsrc, unsigned lds_off, int cols,
                                   int rows, long stride_elems, int padi, int pada) {
  unsigned long long ga = (unsigned long long)(uintptr_t)gsrc;
  v4u g0;
  g0[0] = 1u;                                             // count=1, user descriptor
  g0[1] = lds_off;                                        // LDS byte address
  g0[2] = (unsigned)(ga & 0xFFFFFFFFu);                   // global_addr[31:0]
  g0[3] = (unsigned)((ga >> 32) & 0x01FFFFFFu) | (2u << 30);  // addr[56:32] | type=2
  v8i g1;
  unsigned dw0 = (1u << 16);                              // data_size=1 (2 bytes)
  dw0 |= (1u << 20) | ((unsigned)padi << 22) | ((unsigned)pada << 25);  // pad_enable
  g1[0] = (int)dw0;
  unsigned t0 = (unsigned)cols, t1 = (unsigned)rows;      // tensor dims == tile dims
  g1[1] = (int)((t0 & 0xFFFFu) << 16);                    // tensor_dim0[15:0]
  g1[2] = (int)(((t0 >> 16) & 0xFFFFu) | ((t1 & 0xFFFFu) << 16));
  g1[3] = (int)(((t1 >> 16) & 0xFFFFu) | ((unsigned)cols << 16));  // tile_dim0
  g1[4] = (int)((unsigned)rows & 0xFFFFu);                // tile_dim1 (tile_dim2=0)
  unsigned long long s0 = (unsigned long long)stride_elems;
  g1[5] = (int)(unsigned)(s0 & 0xFFFFFFFFu);              // dim0 stride low
  g1[6] = (int)(unsigned)((s0 >> 32) & 0xFFFFu);          // dim0 stride high
  g1[7] = 0;
  v4i z4 = {0, 0, 0, 0};
#if __clang_major__ >= 23
  v8i z8 = {0, 0, 0, 0, 0, 0, 0, 0};
  __builtin_amdgcn_tensor_load_to_lds(g0, g1, z4, z4, z8, 0);
#else
  __builtin_amdgcn_tensor_load_to_lds(g0, g1, z4, z4, 0);
#endif
}
#endif

// ---------------- weight conversion with transpose: (E x) KxN f32 -> (E x) NxK bf16
__global__ __launch_bounds__(256) void f2bf_t_kernel(const float* __restrict__ src,
                                                     bf16* __restrict__ dst, int K,
                                                     int N, long total) {
  long i = (long)blockIdx.x * 256 + threadIdx.x;
  if (i >= total) return;
  long kn = (long)K * N;
  long e = i / kn, r = i % kn;
  int k = (int)(r / N), n = (int)(r % N);
  dst[e * kn + (long)n * K + k] = f2bf(src[i]);
}

// ---------------- rmsnorm: fp32 row -> bf16 row ----------------
__global__ __launch_bounds__(256) void rmsnorm_kernel(const float* __restrict__ X,
                                                      const float* __restrict__ W,
                                                      bf16* __restrict__ Y) {
  __shared__ float red[8];
  const int t = blockIdx.x;
  const float* xr = X + (long)t * kH;
  float ss = 0.f;
  for (int i = threadIdx.x; i < kH; i += 256) { float v = xr[i]; ss += v * v; }
  ss = redsum16(ss);
  ss += __shfl_xor(ss, 16);
  if ((threadIdx.x & 31) == 0) red[threadIdx.x >> 5] = ss;
  __syncthreads();
  float tot = 0.f;
  for (int w = 0; w < 8; ++w) tot += red[w];
  const float rs = rsqrtf(tot / (float)kH + kEPS);
  for (int i = threadIdx.x; i < kH; i += 256)
    Y[(long)t * kH + i] = f2bf(xr[i] * rs * W[i]);
}

// ---------------- generic WMMA GEMM (plain or expert-grouped) ----------------
// C[row][n] = sum_k A[rowmap(row)][k] * Bt_e[n][k]  (+ addend)
// Bt pre-transposed (N x K). Double-buffered LDS, TDM for B, pipelined A prefetch.
// 8 waves, each computing a 32x64 tile (8 WMMAs per k-step).
static constexpr int BM = 64, BN = 256, BK = 32, KP = 40;

__global__ __launch_bounds__(256) void gemm_kernel(
    const bf16* __restrict__ A, int lda, const bf16* __restrict__ Bt, long strideB,
    float* __restrict__ C, int ldc, const float* __restrict__ addend,
    const int* __restrict__ tile_e, const int* __restrict__ tile_r0,
    const int* __restrict__ tile_mv, const int* __restrict__ perm, int K) {
  __shared__ __align__(16) bf16 As[2][BM][KP];
  __shared__ __align__(16) bf16 Bs[2][BN][KP];

  const int tid = threadIdx.x;
  const int n0 = blockIdx.x * BN;
  int row0, mv;
  const bf16* Bp;
  if (tile_e) {
    mv = tile_mv[blockIdx.y];
    if (mv == 0) return;
    row0 = tile_r0[blockIdx.y];
    Bp = Bt + (long)tile_e[blockIdx.y] * strideB;
  } else {
    row0 = blockIdx.y * BM;
    mv = BM;
    Bp = Bt;
  }

  // Each thread owns one (row, 8-elem chunk) of the A tile; resolve gather row once.
  const int ar = tid >> 2;
  const int ac = (tid & 3) * 8;
  int ri = (ar < mv) ? (perm ? perm[row0 + ar] : (row0 + ar)) : 0;
  if (ri < 0) ri = 0;
  const bf16* Arow = A + (long)ri * lda + ac;

  const int lane = tid & 31;
  const int wid = tid >> 5;
  const int msub = (wid & 1) * 32;
  const int nsub = (wid >> 1) * 64;
  const int lhalf = lane >> 4;
  const int l16 = lane & 15;
  const int w0 = __builtin_amdgcn_readfirstlane(wid);

  const v8f zero8 = {0, 0, 0, 0, 0, 0, 0, 0};
  v8f acc[2][4];
#pragma unroll
  for (int im = 0; im < 2; ++im)
#pragma unroll
    for (int inx = 0; inx < 4; ++inx) acc[im][inx] = zero8;

  const int nIter = K / BK;
  // ---- prologue: stage tile 0 into buffer 0 ----
  {
    uint4 a0r = *(const uint4*)(Arow);
    *(uint4*)&As[0][ar][ac] = a0r;
#ifdef USE_TDM
    if (w0 == 0) {
      tdm_load_2d(Bp + (long)n0 * K, (unsigned)(uintptr_t)&Bs[0][0][0], BK, BN,
                  (long)K, /*padi 16dw*/ 3, /*pada 4dw*/ 3);
      __builtin_amdgcn_s_wait_tensorcnt(0);
    }
#else
#pragma unroll
    for (int j = 0; j < 4; ++j) {
      int task = tid + j * 256;
      int r = task >> 2, c = (task & 3) * 8;
      *(uint4*)&Bs[0][r][c] = *(const uint4*)(Bp + (long)(n0 + r) * K + c);
    }
#endif
  }
  __syncthreads();

  for (int i = 0; i < nIter; ++i) {
    const int buf = i & 1, nbuf = buf ^ 1;
    const bool hasNext = (i + 1 < nIter);
    uint4 aNext;
    if (hasNext) {
      aNext = *(const uint4*)(Arow + (i + 1) * BK);   // prefetch A into registers
#ifdef USE_TDM
      if (w0 == 0)
        tdm_load_2d(Bp + (long)n0 * K + (i + 1) * BK,
                    (unsigned)(uintptr_t)&Bs[nbuf][0][0], BK, BN, (long)K, 3, 3);
#else
#pragma unroll
      for (int j = 0; j < 4; ++j) {
        int task = tid + j * 256;
        int r = task >> 2, c = (task & 3) * 8;
        *(uint4*)&Bs[nbuf][r][c] =
            *(const uint4*)(Bp + (long)(n0 + r) * K + (i + 1) * BK + c);
      }
#endif
    }
    // ---- compute on current buffer: 8 WMMAs ----
    v16bf a0 = ld_frag(&As[buf][msub + l16][lhalf * 16],
                       &As[buf][msub + l16][lhalf * 16 + 8]);
    v16bf a1 = ld_frag(&As[buf][msub + 16 + l16][lhalf * 16],
                       &As[buf][msub + 16 + l16][lhalf * 16 + 8]);
    const bf16* bp0 = &Bs[buf][nsub + l16][lhalf * 16];
    const bf16* bp1 = &Bs[buf][nsub + 16 + l16][lhalf * 16];
    const bf16* bp2 = &Bs[buf][nsub + 32 + l16][lhalf * 16];
    const bf16* bp3 = &Bs[buf][nsub + 48 + l16][lhalf * 16];
    v16bf b0 = ld_frag(bp0, bp0 + 8);
    v16bf b1 = ld_frag(bp1, bp1 + 8);
    v16bf b2 = ld_frag(bp2, bp2 + 8);
    v16bf b3 = ld_frag(bp3, bp3 + 8);
    acc[0][0] = wmma_bf16(a0, b0, acc[0][0]);
    acc[1][0] = wmma_bf16(a1, b0, acc[1][0]);
    acc[0][1] = wmma_bf16(a0, b1, acc[0][1]);
    acc[1][1] = wmma_bf16(a1, b1, acc[1][1]);
    acc[0][2] = wmma_bf16(a0, b2, acc[0][2]);
    acc[1][2] = wmma_bf16(a1, b2, acc[1][2]);
    acc[0][3] = wmma_bf16(a0, b3, acc[0][3]);
    acc[1][3] = wmma_bf16(a1, b3, acc[1][3]);
    if (hasNext) {
      *(uint4*)&As[nbuf][ar][ac] = aNext;
#ifdef USE_TDM
      if (w0 == 0) __builtin_amdgcn_s_wait_tensorcnt(0);  // B(i+1) ready before barrier
#endif
    }
    __syncthreads();
  }

  // ---- epilogue: uniform branch on residual-add, straight-line stores ----
  if (addend) {
#pragma unroll
    for (int im = 0; im < 2; ++im)
#pragma unroll
      for (int inx = 0; inx < 4; ++inx) {
        v8f cc = acc[im][inx];
        const int gcol = n0 + nsub + inx * 16 + l16;
#pragma unroll
        for (int r = 0; r < 8; ++r) {
          int m = msub + im * 16 + r + lhalf * 8;
          long gi = (long)(row0 + m) * ldc + gcol;
          C[gi] = cc[r] + addend[gi];
        }
      }
  } else {
#pragma unroll
    for (int im = 0; im < 2; ++im)
#pragma unroll
      for (int inx = 0; inx < 4; ++inx) {
        v8f cc = acc[im][inx];
        const int gcol = n0 + nsub + inx * 16 + l16;
#pragma unroll
        for (int r = 0; r < 8; ++r) {
          int m = msub + im * 16 + r + lhalf * 8;
          long gi = (long)(row0 + m) * ldc + gcol;
          C[gi] = cc[r];
        }
      }
  }
}

// ---------------- qkv split: fp32 (T x 3072) -> bf16 q/k (B,NH,S,HD) + v dim-major
__global__ __launch_bounds__(256) void qkvsplit_kernel(const float* __restrict__ QKV,
                                                       bf16* __restrict__ Qo,
                                                       bf16* __restrict__ Ko,
                                                       bf16* __restrict__ Vtg) {
  long i = (long)blockIdx.x * 256 + threadIdx.x;  // over kT*kQKVN
  int t = (int)(i / kQKVN), c = (int)(i % kQKVN);
  int b = t >> 10, s = t & 1023;
  int which = c >> 10, cc = c & 1023;
  int hh = cc >> 6, d = cc & 63;
  long bh = (long)b * kNH + hh;
  bf16 v = f2bf(QKV[i]);
  if (which == 0) Qo[(bh * kS + s) * kHD + d] = v;
  else if (which == 1) Ko[(bh * kS + s) * kHD + d] = v;
  else Vtg[(bh * kHD + d) * kS + s] = v;   // [b,h,d,s] dim-major
}

// ---------------- flash attention (8 waves x 16 queries / WG, double-buffered) ----
__global__ __launch_bounds__(256) void flash_kernel(const bf16* __restrict__ Q,
                                                    const bf16* __restrict__ Km,
                                                    const bf16* __restrict__ Vtg,
                                                    const float* __restrict__ mask,
                                                    bf16* __restrict__ O) {
  __shared__ __align__(16) bf16 Kt[2][32][72];   // [key][dim], padded
  __shared__ __align__(16) bf16 Vt[2][64][40];   // [dim][key], padded
  __shared__ __align__(16) bf16 Pl[8][16][40];   // per-wave P tile
  const int tid = threadIdx.x, lane = tid & 31, wv = tid >> 5;
  const int lhalf = lane >> 4, l16 = lane & 15;
  const int b = blockIdx.z, h = blockIdx.y;
  const int q0 = blockIdx.x * 128 + wv * 16;
  const long bh = (long)b * kNH + h;
  const bf16* Qb = Q + (bh * kS + q0) * kHD;
  const bf16* Kb = Km + bh * kS * kHD;
  const bf16* Vb = Vtg + bh * kHD * kS;
  const int w0 = __builtin_amdgcn_readfirstlane(wv);

  v16bf aq0 = ld_frag(Qb + (long)l16 * kHD + lhalf * 16,
                      Qb + (long)l16 * kHD + lhalf * 16 + 8);
  v16bf aq1 = ld_frag(Qb + (long)l16 * kHD + 32 + lhalf * 16,
                      Qb + (long)l16 * kHD + 32 + lhalf * 16 + 8);

  float mrow[8], lrow[8];
#pragma unroll
  for (int r = 0; r < 8; ++r) { mrow[r] = -1e30f; lrow[r] = 0.f; }
  v8f o0 = {0,0,0,0,0,0,0,0}, o1 = {0,0,0,0,0,0,0,0};
  v8f o2 = {0,0,0,0,0,0,0,0}, o3 = {0,0,0,0,0,0,0,0};

  const int nIter = kS / 32;
  // ---- prologue: stage K/V block 0 into buffer 0 ----
#ifdef USE_TDM
  if (w0 == 0) {
    tdm_load_2d(Kb, (unsigned)(uintptr_t)&Kt[0][0][0], kHD, 32, (long)kHD,
                /*padi 32dw*/ 4, /*pada 4dw*/ 3);
    tdm_load_2d(Vb, (unsigned)(uintptr_t)&Vt[0][0][0], 32, kHD, (long)kS,
                /*padi 16dw*/ 3, /*pada 4dw*/ 3);
    __builtin_amdgcn_s_wait_tensorcnt(0);
  }
#else
  {
    int kr = tid >> 3, kc = (tid & 7) * 8;
    *(uint4*)&Kt[0][kr][kc] = *(const uint4*)(Kb + (long)kr * kHD + kc);
    int vr = tid >> 2, vc = (tid & 3) * 8;
    *(uint4*)&Vt[0][vr][vc] = *(const uint4*)(Vb + (long)vr * kS + vc);
  }
#endif
  __syncthreads();

  for (int i = 0; i < nIter; ++i) {
    const int buf = i & 1, nbuf = buf ^ 1;
    const bool hasNext = (i + 1 < nIter);
    const int kb = i * 32;
    if (hasNext) {
      const int kn = kb + 32;
#ifdef USE_TDM
      if (w0 == 0) {
        tdm_load_2d(Kb + (long)kn * kHD, (unsigned)(uintptr_t)&Kt[nbuf][0][0], kHD,
                    32, (long)kHD, 4, 3);
        tdm_load_2d(Vb + kn, (unsigned)(uintptr_t)&Vt[nbuf][0][0], 32, kHD,
                    (long)kS, 3, 3);
      }
#else
      int kr = tid >> 3, kc = (tid & 7) * 8;
      *(uint4*)&Kt[nbuf][kr][kc] = *(const uint4*)(Kb + (long)(kn + kr) * kHD + kc);
      int vr = tid >> 2, vc = (tid & 3) * 8;
      *(uint4*)&Vt[nbuf][vr][vc] = *(const uint4*)(Vb + (long)vr * kS + kn + vc);
#endif
    }

    v8f s0 = {0,0,0,0,0,0,0,0}, s1 = {0,0,0,0,0,0,0,0};
    {
      v16bf b00 = ld_frag(&Kt[buf][l16][lhalf * 16], &Kt[buf][l16][lhalf * 16 + 8]);
      v16bf b01 = ld_frag(&Kt[buf][l16][32 + lhalf * 16], &Kt[buf][l16][32 + lhalf * 16 + 8]);
      v16bf b10 = ld_frag(&Kt[buf][16 + l16][lhalf * 16], &Kt[buf][16 + l16][lhalf * 16 + 8]);
      v16bf b11 = ld_frag(&Kt[buf][16 + l16][32 + lhalf * 16], &Kt[buf][16 + l16][32 + lhalf * 16 + 8]);
      s0 = wmma_bf16(aq0, b00, s0);
      s0 = wmma_bf16(aq1, b01, s0);
      s1 = wmma_bf16(aq0, b10, s1);
      s1 = wmma_bf16(aq1, b11, s1);
    }

    float p0[8], p1[8], alpha[8];
#pragma unroll
    for (int r = 0; r < 8; ++r) {
      int q = q0 + r + lhalf * 8;
      p0[r] = s0[r] * 0.125f + mask[(long)q * kS + kb + l16];
      p1[r] = s1[r] * 0.125f + mask[(long)q * kS + kb + 16 + l16];
    }
#pragma unroll
    for (int r = 0; r < 8; ++r) {
      float rm = redmax16(fmaxf(p0[r], p1[r]));
      float mn = fmaxf(mrow[r], rm);
      alpha[r] = __expf(mrow[r] - mn);
      mrow[r] = mn;
      p0[r] = __expf(p0[r] - mn);
      p1[r] = __expf(p1[r] - mn);
      float rs = redsum16(p0[r] + p1[r]);
      lrow[r] = lrow[r] * alpha[r] + rs;
    }
#pragma unroll
    for (int r = 0; r < 8; ++r) {
      o0[r] *= alpha[r]; o1[r] *= alpha[r]; o2[r] *= alpha[r]; o3[r] *= alpha[r];
    }
#pragma unroll
    for (int r = 0; r < 8; ++r) {
      int M = r + lhalf * 8;
      Pl[wv][M][l16] = f2bf(p0[r]);
      Pl[wv][M][16 + l16] = f2bf(p1[r]);
    }
    wavebar();
    v16bf ap = ld_frag(&Pl[wv][l16][lhalf * 16], &Pl[wv][l16][lhalf * 16 + 8]);
    v16bf bv0 = ld_frag(&Vt[buf][l16][lhalf * 16], &Vt[buf][l16][lhalf * 16 + 8]);
    v16bf bv1 = ld_frag(&Vt[buf][16 + l16][lhalf * 16], &Vt[buf][16 + l16][lhalf * 16 + 8]);
    v16bf bv2 = ld_frag(&Vt[buf][32 + l16][lhalf * 16], &Vt[buf][32 + l16][lhalf * 16 + 8]);
    v16bf bv3 = ld_frag(&Vt[buf][48 + l16][lhalf * 16], &Vt[buf][48 + l16][lhalf * 16 + 8]);
    o0 = wmma_bf16(ap, bv0, o0);
    o1 = wmma_bf16(ap, bv1, o1);
    o2 = wmma_bf16(ap, bv2, o2);
    o3 = wmma_bf16(ap, bv3, o3);
#ifdef USE_TDM
    if (w0 == 0 && hasNext) __builtin_amdgcn_s_wait_tensorcnt(0);
#endif
    __syncthreads();
  }

#pragma unroll
  for (int n = 0; n < 4; ++n) {
    v8f oo = (n == 0) ? o0 : (n == 1) ? o1 : (n == 2) ? o2 : o3;
#pragma unroll
    for (int r = 0; r < 8; ++r) {
      int q = q0 + r + lhalf * 8;
      int d = h * kHD + n * 16 + l16;
      O[(long)(b * kS + q) * (kNH * kHD) + d] = f2bf(oo[r] / lrow[r]);
    }
  }
}

// ---------------- router: logits, softmax, top-2 (one wave per token) ----------------
__global__ __launch_bounds__(256) void router_kernel(const bf16* __restrict__ Hm,
                                                     const float* __restrict__ RW,
                                                     int* __restrict__ ti,
                                                     float* __restrict__ tw) {
  const int lane = threadIdx.x & 31, wv = threadIdx.x >> 5;
  const int t = blockIdx.x * 8 + wv;
  const bf16* hr = Hm + (long)t * kH;
  float acc[kE];
#pragma unroll
  for (int e = 0; e < kE; ++e) acc[e] = 0.f;
  for (int i = lane; i < kH; i += 32) {
    float hv = bf2f(hr[i]);
#pragma unroll
    for (int e = 0; e < kE; ++e) acc[e] += hv * RW[(long)i * kE + e];
  }
#pragma unroll
  for (int e = 0; e < kE; ++e) {
    acc[e] = redsum16(acc[e]);
    acc[e] += __shfl_xor(acc[e], 16);
  }
  float mx = acc[0];
#pragma unroll
  for (int e = 1; e < kE; ++e) mx = fmaxf(mx, acc[e]);
  float p[kE], sum = 0.f;
#pragma unroll
  for (int e = 0; e < kE; ++e) { p[e] = __expf(acc[e] - mx); sum += p[e]; }
#pragma unroll
  for (int e = 0; e < kE; ++e) p[e] /= sum;
  int i0 = 0;
#pragma unroll
  for (int e = 1; e < kE; ++e) if (p[e] > p[i0]) i0 = e;
  int i1 = (i0 == 0) ? 1 : 0;
#pragma unroll
  for (int e = 0; e < kE; ++e) if (e != i0 && p[e] > p[i1]) i1 = e;
  float s = p[i0] + p[i1];
  if (lane == 0) {
    ti[t * 2] = i0; ti[t * 2 + 1] = i1;
    tw[t * 2] = p[i0] / s; tw[t * 2 + 1] = p[i1] / s;
  }
}

// ---------------- deterministic routing tables (single thread, tiny) ----------------
__global__ void route_build_kernel(const int* __restrict__ ti, const float* __restrict__ tw,
                                   int* __restrict__ perm, float* __restrict__ wgt,
                                   int* __restrict__ postk, int* __restrict__ te,
                                   int* __restrict__ tr0, int* __restrict__ tmv) {
  if (threadIdx.x != 0 || blockIdx.x != 0) return;
  int cnt[kE];
  for (int e = 0; e < kE; ++e) cnt[e] = 0;
  for (int t = 0; t < kT; ++t) { cnt[ti[2 * t]]++; cnt[ti[2 * t + 1]]++; }
  int off[kE];
  int cur = 0, nt = 0;
  for (int e = 0; e < kE; ++e) {
    off[e] = cur;
    int full = (cnt[e] + 63) >> 6;
    for (int i = 0; i < full; ++i) {
      te[nt] = e;
      tr0[nt] = cur + i * 64;
      int mv = cnt[e] - i * 64;
      tmv[nt] = mv > 64 ? 64 : mv;
      nt++;
    }
    cur += full * 64;
  }
  for (; nt < kMAXT; ++nt) { te[nt] = 0; tr0[nt] = 0; tmv[nt] = 0; }
  int curs[kE];
  for (int e = 0; e < kE; ++e) curs[e] = off[e];
  for (int t = 0; t < kT; ++t)
    for (int k = 0; k < 2; ++k) {
      int e = ti[2 * t + k];
      int p = curs[e]++;
      perm[p] = t;
      wgt[p] = tw[2 * t + k];
      postk[2 * t + k] = p;
    }
}

// ---------------- silu(g)*u * routing-weight -> bf16 ----------------
__global__ __launch_bounds__(256) void silu_kernel(const float* __restrict__ GU,
                                                   const float* __restrict__ wgt,
                                                   bf16* __restrict__ ACT) {
  long i = (long)blockIdx.x * 256 + threadIdx.x;  // over kNA*kINTER
  long row = i / kINTER;
  int j = (int)(i % kINTER);
  float g = GU[row * (2 * kINTER) + j];
  float u = GU[row * (2 * kINTER) + kINTER + j];
  float a = g / (1.f + __expf(-g)) * u * wgt[row];
  ACT[i] = f2bf(a);
}

// ---------------- final: out = x1 + y[pos0] + y[pos1] ----------------
__global__ __launch_bounds__(256) void final_kernel(const float* __restrict__ X1,
                                                    const float* __restrict__ Y,
                                                    const int* __restrict__ postk,
                                                    float* __restrict__ out) {
  long i = (long)blockIdx.x * 256 + threadIdx.x;  // over kT*kH
  int t = (int)(i >> 10), j = (int)(i & 1023);
  int p0 = postk[t * 2], p1 = postk[t * 2 + 1];
  out[i] = X1[i] + Y[(long)p0 * kH + j] + Y[(long)p1 * kH + j];
}

// ---------------- host orchestration ----------------
extern "C" void kernel_launch(void* const* d_in, const int* in_sizes, int n_in,
                              void* d_out, int out_size, void* d_ws, size_t ws_size,
                              hipStream_t stream) {
  const float* hidden = (const float*)d_in[0];
  const float* mask = (const float*)d_in[1];
  const float* ln1 = (const float*)d_in[2];
  const float* ln2 = (const float*)d_in[3];
  const float* qkvw = (const float*)d_in[4];
  const float* ow = (const float*)d_in[5];
  const float* rw = (const float*)d_in[6];
  const float* gupw = (const float*)d_in[7];
  const float* dww = (const float*)d_in[8];
  float* out = (float*)d_out;

  char* ws = (char*)d_ws;
  size_t cur = 0;
  auto alloc = [&](size_t bytes) -> char* {
    char* p = ws + cur;
    cur = (cur + bytes + 255) & ~(size_t)255;
    return p;
  };

  const size_t nQKVW = (size_t)kH * kQKVN;
  const size_t nOW = (size_t)kH * kH;
  const size_t nGUW = (size_t)kE * kH * 2 * kINTER;
  const size_t nDW = (size_t)kE * kINTER * kH;

  bf16* wQKV = (bf16*)alloc(nQKVW * 2);   // transposed: kQKVN x kH
  bf16* wO = (bf16*)alloc(nOW * 2);       // transposed: kH x kH
  bf16* wGU = (bf16*)alloc(nGUW * 2);     // per-expert transposed: 2*kINTER x kH
  bf16* wDW = (bf16*)alloc(nDW * 2);      // per-expert transposed: kH x kINTER
  bf16* H1 = (bf16*)alloc((size_t)kT * kH * 2);
  float* QKVo = (float*)alloc((size_t)kT * kQKVN * 4);
  bf16* Qb = (bf16*)alloc((size_t)kT * kH * 2);
  bf16* Kb = (bf16*)alloc((size_t)kT * kH * 2);
  bf16* Vtg = (bf16*)alloc((size_t)kT * kH * 2);
  bf16* ATT = (bf16*)alloc((size_t)kT * kH * 2);
  float* X1 = (float*)alloc((size_t)kT * kH * 4);
  bf16* H2 = (bf16*)alloc((size_t)kT * kH * 2);
  int* TI = (int*)alloc((size_t)kT * 2 * 4);
  float* TW = (float*)alloc((size_t)kT * 2 * 4);
  int* PERM = (int*)alloc((size_t)kNA * 4);
  float* WGT = (float*)alloc((size_t)kNA * 4);
  int* POSTK = (int*)alloc((size_t)kT * 2 * 4);
  int* TE = (int*)alloc((size_t)kMAXT * 4);
  int* TR0 = (int*)alloc((size_t)kMAXT * 4);
  int* TMV = (int*)alloc((size_t)kMAXT * 4);
  float* GU = (float*)alloc((size_t)kNA * 2 * kINTER * 4);
  bf16* ACT = (bf16*)alloc((size_t)kNA * kINTER * 2);
  float* Y = (float*)alloc((size_t)kNA * kH * 4);

  // weight conversions fp32 -> bf16 with transpose (N-major for TDM/vector staging)
  f2bf_t_kernel<<<(int)((nQKVW + 255) / 256), 256, 0, stream>>>(qkvw, wQKV, kH,
                                                                kQKVN, (long)nQKVW);
  f2bf_t_kernel<<<(int)((nOW + 255) / 256), 256, 0, stream>>>(ow, wO, kH, kH,
                                                              (long)nOW);
  f2bf_t_kernel<<<(int)((nGUW + 255) / 256), 256, 0, stream>>>(gupw, wGU, kH,
                                                               2 * kINTER, (long)nGUW);
  f2bf_t_kernel<<<(int)((nDW + 255) / 256), 256, 0, stream>>>(dww, wDW, kINTER, kH,
                                                              (long)nDW);

  // attention
  rmsnorm_kernel<<<kT, 256, 0, stream>>>(hidden, ln1, H1);
  gemm_kernel<<<dim3(kQKVN / BN, kT / BM), 256, 0, stream>>>(
      H1, kH, wQKV, 0L, QKVo, kQKVN, nullptr, nullptr, nullptr, nullptr, nullptr, kH);
  qkvsplit_kernel<<<(int)((size_t)kT * kQKVN / 256), 256, 0, stream>>>(QKVo, Qb, Kb,
                                                                       Vtg);
  flash_kernel<<<dim3(kS / 128, kNH, kB), 256, 0, stream>>>(Qb, Kb, Vtg, mask, ATT);
  gemm_kernel<<<dim3(kH / BN, kT / BM), 256, 0, stream>>>(
      ATT, kH, wO, 0L, X1, kH, hidden, nullptr, nullptr, nullptr, nullptr, kH);

  // routed MoE
  rmsnorm_kernel<<<kT, 256, 0, stream>>>(X1, ln2, H2);
  router_kernel<<<kT / 8, 256, 0, stream>>>(H2, rw, TI, TW);
  route_build_kernel<<<1, 1, 0, stream>>>(TI, TW, PERM, WGT, POSTK, TE, TR0, TMV);
  gemm_kernel<<<dim3(2 * kINTER / BN, kMAXT), 256, 0, stream>>>(
      H2, kH, wGU, (long)kH * 2 * kINTER, GU, 2 * kINTER, nullptr, TE, TR0, TMV,
      PERM, kH);
  silu_kernel<<<(int)((size_t)kNA * kINTER / 256), 256, 0, stream>>>(GU, WGT, ACT);
  gemm_kernel<<<dim3(kH / BN, kMAXT), 256, 0, stream>>>(
      ACT, kINTER, wDW, (long)kINTER * kH, Y, kH, nullptr, TE, TR0, TMV, nullptr,
      kINTER);
  final_kernel<<<(int)((size_t)kT * kH / 256), 256, 0, stream>>>(X1, Y, POSTK, out);
}
